// LSTMClassifier_5815385719232
// MI455X (gfx1250) — matrix operand order repaired and checked by reading, not compiled, not run on hardware
//
#include <hip/hip_runtime.h>
#include <stdint.h>

// ---------------------------------------------------------------------------
// LSTMClassifier on gfx1250 (MI455X, wave32, WMMA bf16 16x16x32)
// B=32, S=256, E=256, H=512, 4 layers, NUM_CLASSES=4
//
// Structure:
//  - gx = X @ Wih^T + biases  : whole-chip bf16 WMMA GEMM (per layer)
//  - recurrence               : ONE persistent kernel per layer, 32 WGs.
//      * Whh slice LDS-resident for the whole layer
//      * h_{t-1} async-copied global->LDS each step (ASYNCcnt path)
//      * inner loop is pure ds_load_b128 + v_wmma
//      * cell state in registers
//      * 32-WG grid barrier (global atomics + agent fences) per timestep
//  - softmax over S + tiny FC : fp32 reduction kernels
// ---------------------------------------------------------------------------

#define BB   32
#define SS   256
#define EE   256
#define HH   512
#define G4H  (4 * HH)     // 2048
#define NCLS 4
#define NLAY 4
#define MROWS (BB * SS)   // 8192
#define BPAD 520          // padded LDS row stride (u16) for Whh slice
#define APAD 520          // padded LDS row stride (u16) for h staging

typedef unsigned short u16;
typedef __attribute__((ext_vector_type(16))) __bf16 v16bf;
typedef __attribute__((ext_vector_type(8)))  float  v8f;
typedef __attribute__((ext_vector_type(8)))  unsigned short ushort8;

// ---------------- helpers ----------------

__device__ __forceinline__ u16 f2bf(float f) {
  union { float f; unsigned u; } x; x.f = f;
  unsigned u = x.u;
  u += 0x7fffu + ((u >> 16) & 1u);     // round-to-nearest-even
  return (u16)(u >> 16);
}

__device__ __forceinline__ float sigmoidf(float x) {
  return 1.0f / (1.0f + __expf(-x));
}

// A-matrix per-lane fragment: two contiguous 16B chunks (K and K+8: 16 consecutive K)
__device__ __forceinline__ v16bf load_a16(const u16* p0, const u16* p1) {
  union { ushort8 h[2]; v16bf v; } u;
  u.h[0] = *reinterpret_cast<const ushort8*>(p0);
  u.h[1] = *reinterpret_cast<const ushort8*>(p1);
  return u.v;
}

// B-matrix per-lane fragment: one contiguous 32B chunk (16 consecutive K)
__device__ __forceinline__ v16bf load_b16(const u16* p) {
  union { ushort8 h[2]; v16bf v; } u;
  u.h[0] = *reinterpret_cast<const ushort8*>(p);
  u.h[1] = *reinterpret_cast<const ushort8*>(p + 8);
  return u.v;
}

// async copy of one 16B chunk: global -> LDS (per-lane), tracked by ASYNCcnt.
// VDST = LDS byte offset (low 32 bits of the generic pointer);
// VADDR = 64-bit global address; no SADDR.
__device__ __forceinline__ void async_g2l_b128(const u16* gsrc, const u16* ldst) {
  const unsigned loff = (unsigned)(uintptr_t)ldst;
  asm volatile("global_load_async_to_lds_b128 %0, %1, off"
               :: "v"(loff), "v"(gsrc) : "memory");
}

__device__ __forceinline__ void wait_asynccnt0() {
  asm volatile("s_wait_asynccnt 0x0" ::: "memory");
}

// ---------------- kernels ----------------

__global__ void k_f32_to_bf16(const float* __restrict__ in, u16* __restrict__ out, int n) {
  int i = blockIdx.x * blockDim.x + threadIdx.x;
  if (i < n) out[i] = f2bf(in[i]);
}

// gx = A[MROWS,K] @ W[G4H,K]^T + (bih + bhh)    (bf16 in, f32 out)
// grid: (G4H/64, MROWS/128), block 256 (8 waves). Wave w -> 16(M) x 64(N) strip.
__global__ void k_gemm_gx(const u16* __restrict__ A, const u16* __restrict__ W,
                          const float* __restrict__ bih, const float* __restrict__ bhh,
                          float* __restrict__ C, int K) {
  const int tid    = threadIdx.x;
  const int wave   = tid >> 5;
  const int lane   = tid & 31;
  const int lane16 = lane & 15;
  const int hi     = (lane >> 4) & 1;

  const int m_base = blockIdx.y * 128 + wave * 16;
  const int n_base = blockIdx.x * 64;

  const u16* arow  = A + (size_t)(m_base + lane16) * K + (hi ? 16 : 0);
  const u16* brow0 = W + (size_t)(n_base +  0 + lane16) * K + (hi ? 16 : 0);
  const u16* brow1 = W + (size_t)(n_base + 16 + lane16) * K + (hi ? 16 : 0);
  const u16* brow2 = W + (size_t)(n_base + 32 + lane16) * K + (hi ? 16 : 0);
  const u16* brow3 = W + (size_t)(n_base + 48 + lane16) * K + (hi ? 16 : 0);

  v8f acc0 = {}, acc1 = {}, acc2 = {}, acc3 = {};

  for (int k0 = 0; k0 < K; k0 += 32) {
    v16bf a  = load_a16(arow + k0, arow + k0 + 8);
    v16bf b0 = load_b16(brow0 + k0);
    v16bf b1 = load_b16(brow1 + k0);
    v16bf b2 = load_b16(brow2 + k0);
    v16bf b3 = load_b16(brow3 + k0);
    acc0 = __builtin_amdgcn_wmma_f32_16x16x32_bf16(false, a, false, b0, (short)0, acc0, false, false);
    acc1 = __builtin_amdgcn_wmma_f32_16x16x32_bf16(false, a, false, b1, (short)0, acc1, false, false);
    acc2 = __builtin_amdgcn_wmma_f32_16x16x32_bf16(false, a, false, b2, (short)0, acc2, false, false);
    acc3 = __builtin_amdgcn_wmma_f32_16x16x32_bf16(false, a, false, b3, (short)0, acc3, false, false);
  }

  // D layout: vgpr j -> m = j + 8*hi ; n = lane16
  #pragma unroll
  for (int j = 0; j < 8; ++j) {
    const size_t m = (size_t)(m_base + j + 8 * hi);
    int n;
    n = n_base +  0 + lane16; C[m * G4H + n] = acc0[j] + bih[n] + bhh[n];
    n = n_base + 16 + lane16; C[m * G4H + n] = acc1[j] + bih[n] + bhh[n];
    n = n_base + 32 + lane16; C[m * G4H + n] = acc2[j] + bih[n] + bhh[n];
    n = n_base + 48 + lane16; C[m * G4H + n] = acc3[j] + bih[n] + bhh[n];
  }
}

// Persistent per-layer LSTM recurrence.
// grid 32 WGs (H/16 column groups) x 256 threads (8 waves = 2 M-tiles x 4 gates).
__global__ void k_lstm_layer(const float* __restrict__ gx,
                             const u16* __restrict__ Whh_l,   // [4H, H] bf16
                             float* __restrict__ hseq_f,      // [B,S,H] f32 (final layer)
                             u16*   __restrict__ hseq_b,      // [B,S,H] bf16 (in: t-1, out: t)
                             unsigned* __restrict__ bar,      // zeroed counter for this layer
                             int write_f32) {
  __shared__ u16   B_lds[64 * BPAD];     // 66,560 B: Whh slice, padded rows
  __shared__ u16   A_lds[32 * APAD];     // 33,280 B: h_{t-1} staging, padded rows
  __shared__ float g_lds[4][32][17];     //  8,704 B: gate pre-activations

  const int tid    = threadIdx.x;
  const int wg     = blockIdx.x;
  const int wave   = tid >> 5;
  const int lane   = tid & 31;
  const int lane16 = lane & 15;
  const int hi     = (lane >> 4) & 1;
  const int gate   = wave & 3;
  const int mt     = wave >> 2;

  // ---- stage Whh slice into LDS (once per layer): 64 rows x 512 bf16 ----
  #pragma unroll
  for (int i = 0; i < 16; ++i) {
    const int c  = tid + i * 256;        // 4096 chunks of 8 u16
    const int rb = c >> 6;               // 0..63 (gate*16 + r)
    const int c8 = c & 63;
    const int grow = (rb >> 4) * HH + wg * 16 + (rb & 15);
    const ushort8 v = *reinterpret_cast<const ushort8*>(Whh_l + (size_t)grow * HH + c8 * 8);
    *reinterpret_cast<ushort8*>(&B_lds[rb * BPAD + c8 * 8]) = v;
  }
  __syncthreads();

  const u16* brow   = &B_lds[(gate * 16 + lane16) * BPAD + hi * 16];
  const u16* arow_l = &A_lds[(mt * 16 + lane16) * APAD + hi * 16];

  // per-thread cell state (thread owns fixed (b, h) pairs across all t)
  const int b1 = tid >> 4,         col1 = tid & 15;
  const int b2 = (tid + 256) >> 4, col2 = tid & 15;
  float c1 = 0.0f, c2 = 0.0f;

  for (int t = 0; t < SS; ++t) {
    // ---- async-stage h_{t-1} (32 x 512 bf16) into LDS ----
    if (t > 0) {
      #pragma unroll
      for (int i = 0; i < 8; ++i) {
        const int c   = tid + i * 256;   // 2048 chunks of 8 u16
        const int row = c >> 6;          // batch 0..31
        const int c8  = c & 63;
        const u16* g = hseq_b + ((size_t)row * SS + (t - 1)) * HH + c8 * 8;
        async_g2l_b128(g, &A_lds[row * APAD + c8 * 8]);
      }
      wait_asynccnt0();
    }
    __syncthreads();

    // ---- recurrent GEMM tile: g += h_{t-1} @ Whh^T (all-LDS inner loop) ----
    v8f acc = {};
    if (t > 0) {
      for (int k0 = 0; k0 < HH; k0 += 32) {
        v16bf a = load_a16(arow_l + k0, arow_l + k0 + 8);    // ds_load_b128 x2
        v16bf b = load_b16(brow + k0);                       // ds_load_b128 x2
        acc = __builtin_amdgcn_wmma_f32_16x16x32_bf16(false, a, false, b, (short)0, acc, false, false);
      }
    }
    #pragma unroll
    for (int j = 0; j < 8; ++j)
      g_lds[gate][mt * 16 + j + 8 * hi][lane16] = acc[j];
    __syncthreads();

    // ---- gating: 2 elements per thread, gx read coalesced here ----
    {
      const int hh = wg * 16 + col1;
      const float* gxp = gx + ((size_t)b1 * SS + t) * G4H + hh;
      const float gi = g_lds[0][b1][col1] + gxp[0 * HH];
      const float gf = g_lds[1][b1][col1] + gxp[1 * HH];
      const float gg = g_lds[2][b1][col1] + gxp[2 * HH];
      const float go = g_lds[3][b1][col1] + gxp[3 * HH];
      c1 = sigmoidf(gf) * c1 + sigmoidf(gi) * tanhf(gg);
      const float h = sigmoidf(go) * tanhf(c1);
      const size_t off = ((size_t)b1 * SS + t) * HH + hh;
      hseq_b[off] = f2bf(h);
      if (write_f32) hseq_f[off] = h;
      if (t + 1 < SS) __builtin_prefetch(gxp + G4H, 0, 1);   // next step's gx line
    }
    {
      const int hh = wg * 16 + col2;
      const float* gxp = gx + ((size_t)b2 * SS + t) * G4H + hh;
      const float gi = g_lds[0][b2][col2] + gxp[0 * HH];
      const float gf = g_lds[1][b2][col2] + gxp[1 * HH];
      const float gg = g_lds[2][b2][col2] + gxp[2 * HH];
      const float go = g_lds[3][b2][col2] + gxp[3 * HH];
      c2 = sigmoidf(gf) * c2 + sigmoidf(gi) * tanhf(gg);
      const float h = sigmoidf(go) * tanhf(c2);
      const size_t off = ((size_t)b2 * SS + t) * HH + hh;
      hseq_b[off] = f2bf(h);
      if (write_f32) hseq_f[off] = h;
      if (t + 1 < SS) __builtin_prefetch(gxp + G4H, 0, 1);   // next step's gx line
    }
    __syncthreads();

    // ---- 32-WG grid barrier: h[t] visible before any WG starts step t+1 ----
    if (tid == 0) {
      __builtin_amdgcn_fence(__ATOMIC_RELEASE, "agent");
      atomicAdd(bar, 1u);
      const unsigned target = 32u * (unsigned)(t + 1);
      while (__hip_atomic_load(bar, __ATOMIC_RELAXED, __HIP_MEMORY_SCOPE_AGENT) < target)
        __builtin_amdgcn_s_sleep(1);
    }
    __syncthreads();
    __builtin_amdgcn_fence(__ATOMIC_ACQUIRE, "agent");
  }
}

// softmax over the sequence axis: per (b,h) compute max_s and sum_s exp
__global__ void k_softmax_stats(const float* __restrict__ hseq,
                                float* __restrict__ mx, float* __restrict__ sm) {
  const int idx = blockIdx.x * blockDim.x + threadIdx.x;   // B*H
  if (idx >= BB * HH) return;
  const int b = idx / HH;
  const int h = idx % HH;
  const float* base = hseq + (size_t)b * SS * HH + h;
  float m = -3.402823466e+38f;
  for (int s = 0; s < SS; ++s) m = fmaxf(m, base[(size_t)s * HH]);
  float acc = 0.0f;
  for (int s = 0; s < SS; ++s) acc += __expf(base[(size_t)s * HH] - m);
  mx[idx] = m;
  sm[idx] = acc;
}

// out[b,s,c] = sum_h softmax(h)[b,s,h] * Wfc[c,h] + bfc[c]; one WG per (b,s)
__global__ void k_fc(const float* __restrict__ hseq,
                     const float* __restrict__ mx, const float* __restrict__ sm,
                     const float* __restrict__ Wfc, const float* __restrict__ bfc,
                     float* __restrict__ out) {
  __shared__ float red[NCLS][256];
  const int tid = threadIdx.x;
  const int bs  = blockIdx.x;          // b*S + s
  const int b   = bs / SS;

  float acc[NCLS] = {0.f, 0.f, 0.f, 0.f};
  for (int h = tid; h < HH; h += 256) {
    const float v = hseq[(size_t)bs * HH + h];
    const float p = __expf(v - mx[b * HH + h]) / sm[b * HH + h];
    #pragma unroll
    for (int c = 0; c < NCLS; ++c) acc[c] += p * Wfc[c * HH + h];
  }
  #pragma unroll
  for (int c = 0; c < NCLS; ++c) red[c][tid] = acc[c];
  __syncthreads();
  for (int s = 128; s > 0; s >>= 1) {
    if (tid < s) {
      #pragma unroll
      for (int c = 0; c < NCLS; ++c) red[c][tid] += red[c][tid + s];
    }
    __syncthreads();
  }
  if (tid < NCLS) out[(size_t)bs * NCLS + tid] = red[tid][0] + bfc[tid];
}

// ---------------- host side ----------------

extern "C" void kernel_launch(void* const* d_in, const int* in_sizes, int n_in,
                              void* d_out, int out_size, void* d_ws, size_t ws_size,
                              hipStream_t stream) {
  (void)in_sizes; (void)n_in; (void)out_size; (void)ws_size;

  const float* x        = (const float*)d_in[0];   // [B,S,E]
  const float* Wih0     = (const float*)d_in[1];   // [4H,E]
  const float* Wih_rest = (const float*)d_in[2];   // [3,4H,H]
  const float* Whh      = (const float*)d_in[3];   // [4,4H,H]
  const float* bih      = (const float*)d_in[4];   // [4,4H]
  const float* bhh      = (const float*)d_in[5];   // [4,4H]
  const float* Wfc      = (const float*)d_in[6];   // [C,H]
  const float* bfc      = (const float*)d_in[7];   // [C]
  float* out = (float*)d_out;

  // workspace carve-out (~113 MB), 256B aligned slices
  char* p = (char*)d_ws;
  auto carve = [&](size_t bytes) { void* r = (void*)p; p += (bytes + 255) & ~(size_t)255; return r; };
  u16*      x_bf    = (u16*)     carve((size_t)BB * SS * EE * 2);
  u16*      wih0_bf = (u16*)     carve((size_t)G4H * EE * 2);
  u16*      wihr_bf = (u16*)     carve((size_t)3 * G4H * HH * 2);
  u16*      whh_bf  = (u16*)     carve((size_t)NLAY * G4H * HH * 2);
  float*    gx      = (float*)   carve((size_t)MROWS * G4H * 4);
  float*    hseq_f  = (float*)   carve((size_t)BB * SS * HH * 4);
  u16*      hseq_b  = (u16*)     carve((size_t)BB * SS * HH * 2);
  unsigned* bar     = (unsigned*)carve(NLAY * sizeof(unsigned));
  float*    mx      = (float*)   carve((size_t)BB * HH * 4);
  float*    sm      = (float*)   carve((size_t)BB * HH * 4);

  // 1) convert activations/weights to bf16; zero the per-layer barrier counters
  {
    int n;
    n = BB * SS * EE;      k_f32_to_bf16<<<(n + 255) / 256, 256, 0, stream>>>(x, x_bf, n);
    n = G4H * EE;          k_f32_to_bf16<<<(n + 255) / 256, 256, 0, stream>>>(Wih0, wih0_bf, n);
    n = 3 * G4H * HH;      k_f32_to_bf16<<<(n + 255) / 256, 256, 0, stream>>>(Wih_rest, wihr_bf, n);
    n = NLAY * G4H * HH;   k_f32_to_bf16<<<(n + 255) / 256, 256, 0, stream>>>(Whh, whh_bf, n);
  }
  hipMemsetAsync(bar, 0, NLAY * sizeof(unsigned), stream);

  // 2) layers: big gx GEMM + one persistent recurrence kernel each
  for (int l = 0; l < NLAY; ++l) {
    const u16* A = (l == 0) ? x_bf : hseq_b;
    const int  K = (l == 0) ? EE : HH;
    const u16* W = (l == 0) ? wih0_bf : (wihr_bf + (size_t)(l - 1) * G4H * HH);

    k_gemm_gx<<<dim3(G4H / 64, MROWS / 128), 256, 0, stream>>>(
        A, W, bih + (size_t)l * G4H, bhh + (size_t)l * G4H, gx, K);

    k_lstm_layer<<<HH / 16, 256, 0, stream>>>(
        gx, whh_bf + (size_t)l * G4H * HH, hseq_f, hseq_b, bar + l,
        (l == NLAY - 1) ? 1 : 0);
  }

  // 3) softmax over S, then FC
  k_softmax_stats<<<(BB * HH + 255) / 256, 256, 0, stream>>>(hseq_f, mx, sm);
  k_fc<<<BB * SS, 256, 0, stream>>>(hseq_f, mx, sm, Wfc, bfc, out);
}
